// AttentionLayer_89936615178416
// MI455X (gfx1250) — matrix operand order repaired and checked by reading, not compiled, not run on hardware
//
#include <hip/hip_runtime.h>
#include <hip/hip_bf16.h>

typedef __bf16 bf16_t;
typedef __attribute__((ext_vector_type(16))) __bf16 v16bf;
typedef __attribute__((ext_vector_type(8)))  __bf16 v8bf;
typedef __attribute__((ext_vector_type(8)))  float  v8f;

static __device__ inline v8f wmma_bf16(v16bf a, v16bf b, v8f c) {
  // D = A(16x32 bf16) * B(32x16 bf16) + C(16x16 f32)
  return __builtin_amdgcn_wmma_f32_16x16x32_bf16(
      /*neg_a=*/false, a, /*neg_b=*/false, b,
      /*c_mod=*/(short)0, c, /*reuse_a=*/false, /*reuse_b=*/false);
}

// 16 contiguous bf16 -> B-fragment (lane covers k = 16h .. 16h+15)
static __device__ inline v16bf ld16(const bf16_t* p) {
  v8bf a = *(const v8bf*)p;
  v8bf b = *(const v8bf*)(p + 8);
  return __builtin_shufflevector(a, b, 0,1,2,3,4,5,6,7,8,9,10,11,12,13,14,15);
}

// A-fragment from a row pointer: strips at k=8h..8h+7 and k=16+8h..16+8h+7
static __device__ inline v16bf ldA(const bf16_t* rowp, int h) {
  v8bf a = *(const v8bf*)(rowp + 8 * h);
  v8bf b = *(const v8bf*)(rowp + 16 + 8 * h);
  return __builtin_shufflevector(a, b, 0,1,2,3,4,5,6,7,8,9,10,11,12,13,14,15);
}

static __device__ inline v8bf cvt8(float4 a, float4 b) {
  v8bf r;
  r[0] = (bf16_t)a.x; r[1] = (bf16_t)a.y; r[2] = (bf16_t)a.z; r[3] = (bf16_t)a.w;
  r[4] = (bf16_t)b.x; r[5] = (bf16_t)b.y; r[6] = (bf16_t)b.z; r[7] = (bf16_t)b.w;
  return r;
}

// ---------------------------------------------------------------------------
// GEMM: [8192,1024] x [1024,1024] + bias, 128x128 block tile, 8 waves,
// each wave = 64x32 = 8 WMMA C-tiles. K-step 32 (one WMMA K).
// MODE 0: out f32 row-major (qp)         MODE 1: out bf16 [B,H,S,DK] (kp)
// MODE 2: out bf16 [B,H,DV,S] (vpT)      MODE 3: bf16 A, out f32 + residual (x)
// ---------------------------------------------------------------------------
template <int MODE>
__global__ __launch_bounds__(256) void gemm128(const void* __restrict__ Ap,
                                               const float* __restrict__ W,
                                               const float* __restrict__ bias,
                                               const float* __restrict__ resid,
                                               float* __restrict__ outF,
                                               bf16_t* __restrict__ outB) {
  constexpr int Kd = 1024, N = 1024, LDT = 40;
  __shared__ alignas(16) bf16_t sA[128 * LDT];
  __shared__ alignas(16) bf16_t sB[128 * LDT];

  const int tid  = threadIdx.x;
  const int lane = tid & 31, wave = tid >> 5;
  const int h    = lane >> 4, ln = lane & 15;
  const int wm   = wave >> 2, wn = wave & 3;
  const int bM   = blockIdx.x * 128, bN = blockIdx.y * 128;

  v8f acc[4][2] = {};

  for (int kb = 0; kb < Kd / 32; ++kb) {
    if constexpr (MODE == 3) {
      const bf16_t* A = (const bf16_t*)Ap;
#pragma unroll
      for (int i = 0; i < 2; ++i) {
        int seg = tid + 256 * i;
        int row = seg >> 2, kc = (seg & 3) * 8;
        v8bf d = *(const v8bf*)(A + (size_t)(bM + row) * Kd + kb * 32 + kc);
#pragma unroll
        for (int j = 0; j < 8; ++j) sA[row * LDT + kc + j] = d[j];
      }
    } else {
      const float* A = (const float*)Ap;
#pragma unroll
      for (int i = 0; i < 4; ++i) {
        int seg = tid + 256 * i;
        int row = seg >> 3, kc = (seg & 7) * 4;
        float4 d = *(const float4*)(A + (size_t)(bM + row) * Kd + kb * 32 + kc);
        sA[row * LDT + kc + 0] = (bf16_t)d.x;
        sA[row * LDT + kc + 1] = (bf16_t)d.y;
        sA[row * LDT + kc + 2] = (bf16_t)d.z;
        sA[row * LDT + kc + 3] = (bf16_t)d.w;
      }
    }
    // W tile transposed into LDS: sB[n][k]
#pragma unroll
    for (int i = 0; i < 4; ++i) {
      int seg = tid + 256 * i;
      int kk = seg >> 5, nc = (seg & 31) * 4;
      float4 d = *(const float4*)(W + (size_t)(kb * 32 + kk) * N + bN + nc);
      sB[(nc + 0) * LDT + kk] = (bf16_t)d.x;
      sB[(nc + 1) * LDT + kk] = (bf16_t)d.y;
      sB[(nc + 2) * LDT + kk] = (bf16_t)d.z;
      sB[(nc + 3) * LDT + kk] = (bf16_t)d.w;
    }
    __syncthreads();

    v16bf af[4], bfr[2];
#pragma unroll
    for (int mf = 0; mf < 4; ++mf)
      af[mf] = ldA(sA + (wm * 64 + mf * 16 + ln) * LDT, h);
#pragma unroll
    for (int nf = 0; nf < 2; ++nf)
      bfr[nf] = ldA(sB + (wn * 32 + nf * 16 + ln) * LDT, h);
#pragma unroll
    for (int mf = 0; mf < 4; ++mf)
#pragma unroll
      for (int nf = 0; nf < 2; ++nf)
        acc[mf][nf] = wmma_bf16(af[mf], bfr[nf], acc[mf][nf]);
    __syncthreads();
  }

#pragma unroll
  for (int mf = 0; mf < 4; ++mf) {
#pragma unroll
    for (int nf = 0; nf < 2; ++nf) {
      int col = bN + wn * 32 + nf * 16 + ln;
      float bi = bias[col];
#pragma unroll
      for (int v = 0; v < 8; ++v) {
        int row = bM + wm * 64 + mf * 16 + v + 8 * h;
        float val = acc[mf][nf][v] + bi;
        if constexpr (MODE == 0) {
          outF[(size_t)row * N + col] = val;
        } else if constexpr (MODE == 1) {
          int b = row >> 11, s = row & 2047, hd = col >> 6, dk = col & 63;
          outB[(((size_t)(b * 16 + hd)) * 2048 + s) * 64 + dk] = (bf16_t)val;
        } else if constexpr (MODE == 2) {
          int b = row >> 11, s = row & 2047, hd = col >> 6, dv = col & 63;
          outB[(((size_t)(b * 16 + hd)) * 64 + dv) * 2048 + s] = (bf16_t)val;
        } else {
          outF[(size_t)row * N + col] = val + resid[(size_t)row * N + col];
        }
      }
    }
  }
}

// ---------------------------------------------------------------------------
// Flash attention: 4 waves/block, each wave owns 16 query rows (Q-block 64).
// grid = (S/64, B*H). Q kept in registers; K/V B-frags loaded from global
// (L2-resident); P re-laid out C->A through per-wave LDS + s_wait_dscnt.
// ---------------------------------------------------------------------------
__global__ __launch_bounds__(128) void flash_attn(const float* __restrict__ qp,
                                                  const bf16_t* __restrict__ kp,
                                                  const bf16_t* __restrict__ vpT,
                                                  bf16_t* __restrict__ ctx) {
  constexpr int S = 2048, Dm = 1024, H = 16;
  __shared__ alignas(16) bf16_t pbuf[4][16 * 40];

  const int tid = threadIdx.x, lane = tid & 31, wave = tid >> 5;
  const int h2 = lane >> 4, ln = lane & 15;
  const int bh = blockIdx.y, b = bh >> 4, hd = bh & 15;
  const int q0 = blockIdx.x * 64 + wave * 16;

  // Q A-fragments: two 16x32 chunks of DK=64, converted f32->bf16
  const float* qbase = qp + ((size_t)(b * S + q0 + ln)) * Dm + hd * 64;
  v16bf qf[2];
#pragma unroll
  for (int c = 0; c < 2; ++c) {
    const float* p0 = qbase + c * 32 + 8 * h2;
    const float* p1 = qbase + c * 32 + 16 + 8 * h2;
    float4 a0 = *(const float4*)p0, a1 = *(const float4*)(p0 + 4);
    float4 b0 = *(const float4*)p1, b1 = *(const float4*)(p1 + 4);
    qf[c] = __builtin_shufflevector(cvt8(a0, a1), cvt8(b0, b1),
                                    0,1,2,3,4,5,6,7,8,9,10,11,12,13,14,15);
  }

  const bf16_t* kb_ = kp + (size_t)(b * H + hd) * S * 64;   // [key][dk]
  const bf16_t* vb_ = vpT + (size_t)(b * H + hd) * 64 * S;  // [dv][key]

  float m[8], l[8];
  v8f o[4] = {};
#pragma unroll
  for (int v = 0; v < 8; ++v) { m[v] = -3.0e38f; l[v] = 0.f; }

  bf16_t* pw = &pbuf[wave][0];

  for (int kt = 0; kt < S / 32; ++kt) {
    const int key0 = kt * 32;
    v8f s0 = {}, s1 = {};
#pragma unroll
    for (int c = 0; c < 2; ++c) {
      v16bf k0 = ldA(kb_ + (size_t)(key0 + ln) * 64 + c * 32, h2);
      v16bf k1 = ldA(kb_ + (size_t)(key0 + 16 + ln) * 64 + c * 32, h2);
      s0 = wmma_bf16(qf[c], k0, s0);
      s1 = wmma_bf16(qf[c], k1, s1);
    }
    float mt[8], rs[8], sc[8];
#pragma unroll
    for (int v = 0; v < 8; ++v) {
      s0[v] *= 0.125f; s1[v] *= 0.125f;           // 1/sqrt(DK)
      mt[v] = fmaxf(s0[v], s1[v]);
    }
#pragma unroll
    for (int off = 1; off < 16; off <<= 1)        // stays within 16-lane half
#pragma unroll
      for (int v = 0; v < 8; ++v) mt[v] = fmaxf(mt[v], __shfl_xor(mt[v], off, 32));
#pragma unroll
    for (int v = 0; v < 8; ++v) {
      float mn = fmaxf(m[v], mt[v]);
      sc[v] = __expf(m[v] - mn);
      m[v] = mn;
      float p0 = __expf(s0[v] - mn), p1 = __expf(s1[v] - mn);
      s0[v] = p0; s1[v] = p1; rs[v] = p0 + p1;
    }
#pragma unroll
    for (int off = 1; off < 16; off <<= 1)
#pragma unroll
      for (int v = 0; v < 8; ++v) rs[v] += __shfl_xor(rs[v], off, 32);
#pragma unroll
    for (int v = 0; v < 8; ++v) l[v] = l[v] * sc[v] + rs[v];
#pragma unroll
    for (int t = 0; t < 4; ++t)
#pragma unroll
      for (int v = 0; v < 8; ++v) o[t][v] *= sc[v];

    // P: C-layout -> LDS [row][key] (bf16), then reload as A-fragment
#pragma unroll
    for (int v = 0; v < 8; ++v) {
      int rl = v + 8 * h2;
      pw[rl * 40 + ln]      = (bf16_t)s0[v];
      pw[rl * 40 + 16 + ln] = (bf16_t)s1[v];
    }
    asm volatile("s_wait_dscnt 0" ::: "memory");
    v16bf pf = ldA(pw + ln * 40, h2);

#pragma unroll
    for (int t = 0; t < 4; ++t) {
      v16bf vf = ldA(vb_ + (size_t)(t * 16 + ln) * S + key0, h2);
      o[t] = wmma_bf16(pf, vf, o[t]);
    }
  }

#pragma unroll
  for (int v = 0; v < 8; ++v) l[v] = 1.0f / l[v];
#pragma unroll
  for (int t = 0; t < 4; ++t)
#pragma unroll
    for (int v = 0; v < 8; ++v) {
      int s = q0 + v + 8 * h2;
      ctx[((size_t)(b * S + s)) * Dm + hd * 64 + t * 16 + ln] =
          (bf16_t)(o[t][v] * l[v]);
    }
}

// ---------------------------------------------------------------------------
// Rowwise LayerNorm over D=1024, one 256-thread block per row.
// ---------------------------------------------------------------------------
__global__ __launch_bounds__(256) void layernorm_k(const float* __restrict__ x,
                                                   const float* __restrict__ gamma,
                                                   const float* __restrict__ beta,
                                                   float* __restrict__ out) {
  const int row = blockIdx.x, t = threadIdx.x;
  const int lane = t & 31, wave = t >> 5;
  const float4 v = ((const float4*)(x + (size_t)row * 1024))[t];
  float s = v.x + v.y + v.z + v.w;
  float q = v.x * v.x + v.y * v.y + v.z * v.z + v.w * v.w;
#pragma unroll
  for (int off = 1; off < 32; off <<= 1) {
    s += __shfl_xor(s, off, 32);
    q += __shfl_xor(q, off, 32);
  }
  __shared__ float sh[16];
  if (lane == 0) { sh[wave] = s; sh[8 + wave] = q; }
  __syncthreads();
  float ts = 0.f, tq = 0.f;
#pragma unroll
  for (int w = 0; w < 8; ++w) { ts += sh[w]; tq += sh[8 + w]; }
  float mu  = ts * (1.0f / 1024.0f);
  float var = tq * (1.0f / 1024.0f) - mu * mu;
  float r   = rsqrtf(var + 1e-5f);
  float4 g = ((const float4*)gamma)[t], bb = ((const float4*)beta)[t];
  float4 ov;
  ov.x = (v.x - mu) * r * g.x + bb.x;
  ov.y = (v.y - mu) * r * g.y + bb.y;
  ov.z = (v.z - mu) * r * g.z + bb.z;
  ov.w = (v.w - mu) * r * g.w + bb.w;
  ((float4*)(out + (size_t)row * 1024))[t] = ov;
}

// ---------------------------------------------------------------------------
extern "C" void kernel_launch(void* const* d_in, const int* in_sizes, int n_in,
                              void* d_out, int out_size, void* d_ws, size_t ws_size,
                              hipStream_t stream) {
  (void)in_sizes; (void)n_in; (void)out_size; (void)ws_size;
  const float* q     = (const float*)d_in[0];
  const float* k     = (const float*)d_in[1];
  const float* v     = (const float*)d_in[2];
  const float* Wq    = (const float*)d_in[3];
  const float* bq    = (const float*)d_in[4];
  const float* Wk    = (const float*)d_in[5];
  const float* bk    = (const float*)d_in[6];
  const float* Wv    = (const float*)d_in[7];
  const float* bv    = (const float*)d_in[8];
  const float* Wfc   = (const float*)d_in[9];
  const float* bfc   = (const float*)d_in[10];
  const float* gamma = (const float*)d_in[11];
  const float* beta  = (const float*)d_in[12];

  // Workspace layout (80 MB used): qp f32 32MB | kp bf16 16MB | vpT bf16 16MB |
  // ctx bf16 16MB.  x (f32, 32MB) aliases kp+vpT (dead after attention).
  constexpr size_t QP_B = 8192ull * 1024ull * 4ull;   // 32 MB
  constexpr size_t KP_B = 8192ull * 1024ull * 2ull;   // 16 MB
  char* w = (char*)d_ws;
  float*  qp  = (float*)w;
  bf16_t* kp  = (bf16_t*)(w + QP_B);
  bf16_t* vpT = (bf16_t*)(w + QP_B + KP_B);
  bf16_t* ctx = (bf16_t*)(w + QP_B + 2 * KP_B);
  float*  x   = (float*)(w + QP_B);                   // alias over kp/vpT

  dim3 gg(64, 8), gb(256);
  gemm128<0><<<gg, gb, 0, stream>>>(q, Wq, bq, nullptr, qp, nullptr);
  gemm128<1><<<gg, gb, 0, stream>>>(k, Wk, bk, nullptr, nullptr, kp);
  gemm128<2><<<gg, gb, 0, stream>>>(v, Wv, bv, nullptr, nullptr, vpT);
  flash_attn<<<dim3(32, 64), 128, 0, stream>>>(qp, kp, vpT, ctx);
  gemm128<3><<<gg, gb, 0, stream>>>(ctx, Wfc, bfc, qp, x, nullptr);
  layernorm_k<<<dim3(8192), 256, 0, stream>>>(x, gamma, beta, (float*)d_out);
}
